// KNNMapper_63290638074431
// MI455X (gfx1250) — matrix operand order repaired and checked by reading, not compiled, not run on hardware
//
#include <hip/hip_runtime.h>
#include <hip/hip_bf16.h>

typedef __attribute__((ext_vector_type(16))) __bf16   v16bf;
typedef __attribute__((ext_vector_type(8)))  float    v8f;
typedef __attribute__((ext_vector_type(4)))  unsigned int u32x4;

#define D       512          // feature dim
#define KTOP    16
#define KCHUNKS (D / 32)     // 16 bf16-wmma K-steps per dot product

// ---- float -> bf16 (round to nearest even) ----
__device__ __forceinline__ unsigned short f2bf(float f) {
    unsigned int u = __float_as_uint(f);
    u = u + 0x7FFFu + ((u >> 16) & 1u);
    return (unsigned short)(u >> 16);
}

// ---------------------------------------------------------------------------
// Kernel 1: L2-normalize each row of x and store as bf16.
// ---------------------------------------------------------------------------
__global__ __launch_bounds__(128) void normalize_x_kernel(
    const float* __restrict__ x, unsigned short* __restrict__ xbf) {
    __shared__ float red[128];
    const int row = blockIdx.x;
    const int t   = threadIdx.x;
    const float4 v = ((const float4*)(x + (size_t)row * D))[t];
    red[t] = v.x * v.x + v.y * v.y + v.z * v.z + v.w * v.w;
    __syncthreads();
    for (int s = 64; s > 0; s >>= 1) {
        if (t < s) red[t] += red[t + s];
        __syncthreads();
    }
    const float scale = 1.0f / fmaxf(sqrtf(red[0]), 1e-12f);
    uint2 o;
    o.x = (unsigned int)f2bf(v.x * scale) | ((unsigned int)f2bf(v.y * scale) << 16);
    o.y = (unsigned int)f2bf(v.z * scale) | ((unsigned int)f2bf(v.w * scale) << 16);
    ((uint2*)(xbf + (size_t)row * D))[t] = o;
}

// ---------------------------------------------------------------------------
// Kernel 2: refs f32 -> bf16 (already unit-norm). 64MB result is L2-resident.
// ---------------------------------------------------------------------------
__global__ __launch_bounds__(256) void convert_refs_kernel(
    const float* __restrict__ refs, unsigned short* __restrict__ rbf, int n4) {
    int i = blockIdx.x * blockDim.x + threadIdx.x;
    if (i >= n4) return;
    const float4 v = ((const float4*)refs)[i];
    uint2 o;
    o.x = (unsigned int)f2bf(v.x) | ((unsigned int)f2bf(v.y) << 16);
    o.y = (unsigned int)f2bf(v.z) | ((unsigned int)f2bf(v.w) << 16);
    ((uint2*)rbf)[i] = o;
}

// ---------------------------------------------------------------------------
// Kernel 3: WMMA GEMM + fused per-row top-16.
// Block = 256 threads = 8 waves, one block per 16 x-rows.
// A (SRC0) = refs tile, B (SRC1) = x tile -> lane owns one x row with 8
// candidate scores per WMMA in registers (register-resident top-16).
// - double-buffered 256-feature half-tiles (16x global_load_b128 in flight
//   behind 8 WMMAs), fenced with sched_barrier
// - 4 independent accumulator chains so WMMAs issue back-to-back
// - running pointer advanced by the 128KB tile stride (no address rebuild)
// ---------------------------------------------------------------------------
__global__ __launch_bounds__(256) void knn_topk_kernel(
    const unsigned short* __restrict__ xbf,
    const unsigned short* __restrict__ refsbf,
    float* __restrict__ out, int ntiles_per_wave) {
    __shared__ float sAll[16][257];   // pad 257: conflict-free row scans

    const int tid  = threadIdx.x;
    const int wave = tid >> 5;
    const int lane = tid & 31;
    const int r    = lane & 15;       // row-in-tile (B: N index; A: M index)
    const int h    = lane >> 4;       // half of wave
    const int rowBase = blockIdx.x * 16;

    // B fragments (x rows) resident in VGPRs: lane holds K = h*16 .. h*16+15
    v16bf xfrag[KCHUNKS];
    {
        const unsigned short* xr = xbf + (size_t)(rowBase + r) * D + h * 16;
        #pragma unroll
        for (int kc = 0; kc < KCHUNKS; ++kc) {
            union { u32x4 q[2]; v16bf v; } tb;
            tb.q[0] = *(const u32x4*)(xr + kc * 32);
            tb.q[1] = *(const u32x4*)(xr + kc * 32 + 8);
            xfrag[kc] = tb.v;
        }
    }

    // A half-tile loader from a per-lane pointer: 8 chunks x (2 x b128).
    v16bf buf0[8], buf1[8];
    auto load_half = [&](v16bf* __restrict__ buf, const unsigned short* rr) {
        #pragma unroll
        for (int j = 0; j < 8; ++j) {
            union { u32x4 q[2]; v16bf v; } tb;
            tb.q[0] = *(const u32x4*)(rr + j * 32);
            tb.q[1] = *(const u32x4*)(rr + j * 32 + 8);
            buf[j] = tb.v;
        }
    };

    // per-lane top-16 largest dot products (== 16 smallest distances)
    float best[KTOP];
    #pragma unroll
    for (int i = 0; i < KTOP; ++i) best[i] = -1e30f;
    float thr = -1e30f;

    // running pointer: wave's first tile, this lane's ref row + k-phase
    const size_t TILE_STRIDE = (size_t)8 * 16 * D;     // 8 tiles/step, elements
    const unsigned short* p0 =
        refsbf + (size_t)(wave * 16 + r) * D + h * 16;

    load_half(buf0, p0);

    for (int t = 0; t < ntiles_per_wave; ++t) {
        v8f c0 = {0.f,0.f,0.f,0.f,0.f,0.f,0.f,0.f};
        v8f c1 = c0, c2 = c0, c3 = c0;

        // prefetch second half of this tile while computing on first half
        __builtin_amdgcn_sched_barrier(0);
        load_half(buf1, p0 + 256);
        __builtin_amdgcn_sched_barrier(0);
        c0 = __builtin_amdgcn_wmma_f32_16x16x32_bf16(false, buf0[0], false, xfrag[0], (short)0, c0, false, false);
        c1 = __builtin_amdgcn_wmma_f32_16x16x32_bf16(false, buf0[1], false, xfrag[1], (short)0, c1, false, false);
        c2 = __builtin_amdgcn_wmma_f32_16x16x32_bf16(false, buf0[2], false, xfrag[2], (short)0, c2, false, false);
        c3 = __builtin_amdgcn_wmma_f32_16x16x32_bf16(false, buf0[3], false, xfrag[3], (short)0, c3, false, false);
        c0 = __builtin_amdgcn_wmma_f32_16x16x32_bf16(false, buf0[4], false, xfrag[4], (short)0, c0, false, false);
        c1 = __builtin_amdgcn_wmma_f32_16x16x32_bf16(false, buf0[5], false, xfrag[5], (short)0, c1, false, false);
        c2 = __builtin_amdgcn_wmma_f32_16x16x32_bf16(false, buf0[6], false, xfrag[6], (short)0, c2, false, false);
        c3 = __builtin_amdgcn_wmma_f32_16x16x32_bf16(false, buf0[7], false, xfrag[7], (short)0, c3, false, false);

        // prefetch first half of next tile while computing on second half
        const unsigned short* pn = (t + 1 < ntiles_per_wave) ? (p0 + TILE_STRIDE) : p0;
        __builtin_amdgcn_sched_barrier(0);
        load_half(buf0, pn);
        __builtin_amdgcn_sched_barrier(0);
        c0 = __builtin_amdgcn_wmma_f32_16x16x32_bf16(false, buf1[0], false, xfrag[8],  (short)0, c0, false, false);
        c1 = __builtin_amdgcn_wmma_f32_16x16x32_bf16(false, buf1[1], false, xfrag[9],  (short)0, c1, false, false);
        c2 = __builtin_amdgcn_wmma_f32_16x16x32_bf16(false, buf1[2], false, xfrag[10], (short)0, c2, false, false);
        c3 = __builtin_amdgcn_wmma_f32_16x16x32_bf16(false, buf1[3], false, xfrag[11], (short)0, c3, false, false);
        c0 = __builtin_amdgcn_wmma_f32_16x16x32_bf16(false, buf1[4], false, xfrag[12], (short)0, c0, false, false);
        c1 = __builtin_amdgcn_wmma_f32_16x16x32_bf16(false, buf1[5], false, xfrag[13], (short)0, c1, false, false);
        c2 = __builtin_amdgcn_wmma_f32_16x16x32_bf16(false, buf1[6], false, xfrag[14], (short)0, c2, false, false);
        c3 = __builtin_amdgcn_wmma_f32_16x16x32_bf16(false, buf1[7], false, xfrag[15], (short)0, c3, false, false);
        p0 = pn;

        // merge the 4 chains, then threshold-guarded top-16 insert
        const v8f c = (c0 + c1) + (c2 + c3);
        #pragma unroll
        for (int v = 0; v < 8; ++v) {
            const float s = c[v];
            if (s > thr) {
                bool done = false;
                #pragma unroll
                for (int i = 0; i < KTOP; ++i)
                    if (!done && best[i] == thr) { best[i] = s; done = true; }
                float m = best[0];
                #pragma unroll
                for (int i = 1; i < KTOP; ++i) m = fminf(m, best[i]);
                thr = m;
            }
        }
    }

    // merge 8 waves x 2 halves -> 256 partial-best values per row
    const int col = (wave * 2 + h) * 16;
    #pragma unroll
    for (int i = 0; i < KTOP; ++i) sAll[r][col + i] = best[i];
    __syncthreads();

    if (tid < 16) {
        float sel[KTOP];
        for (int k = 0; k < KTOP; ++k) {          // descending dot = ascending dist
            float m = -1e30f; int mi = 0;
            for (int j = 0; j < 256; ++j) {
                const float v = sAll[tid][j];
                if (v > m) { m = v; mi = j; }
            }
            sAll[tid][mi] = -2e30f;
            sel[k] = m;
        }
        float w[KTOP], sum = 0.f;
        #pragma unroll
        for (int k = 0; k < KTOP; ++k) {
            const float d2 = fmaxf(2.0f - 2.0f * sel[k], 1e-12f);
            w[k] = __expf(-sqrtf(d2));            // SIGMA = 1
            sum += w[k];
        }
        const float inv = 1.0f / fmaxf(sum, 1e-12f);
        float* orow = out + (size_t)(rowBase + tid) * KTOP;
        #pragma unroll
        for (int k = 0; k < KTOP; ++k) orow[k] = w[k] * inv;
    }
}

extern "C" void kernel_launch(void* const* d_in, const int* in_sizes, int n_in,
                              void* d_out, int out_size, void* d_ws, size_t ws_size,
                              hipStream_t stream) {
    const float* x    = (const float*)d_in[0];
    const float* refs = (const float*)d_in[1];
    const int N = in_sizes[0] / D;        // 4096
    const int M = in_sizes[1] / D;        // 65536

    unsigned short* xbf = (unsigned short*)d_ws;            // N*D bf16 (4 MB)
    unsigned short* rbf = xbf + (size_t)N * D;              // M*D bf16 (64 MB)

    normalize_x_kernel<<<N, 128, 0, stream>>>(x, xbf);

    const int n4 = (M * D) / 4;
    convert_refs_kernel<<<(n4 + 255) / 256, 256, 0, stream>>>(refs, rbf, n4);

    knn_topk_kernel<<<N / 16, 256, 0, stream>>>(xbf, rbf, (float*)d_out,
                                                M / (8 * 16));
}
